// QuantumEyeDiseaseClassifier_66778151518506
// MI455X (gfx1250) — matrix operand order repaired and checked by reading, compiled, not run on hardware
//
#include <hip/hip_runtime.h>
#include <hip/hip_bf16.h>
#include <math.h>

typedef __attribute__((ext_vector_type(16))) _Float16 v16h;
typedef __attribute__((ext_vector_type(8)))  _Float16 v8h;
typedef __attribute__((ext_vector_type(8)))  float    v8f;

#define BATCH 32

// ---------------------------------------------------------------------------
// Utility kernels
// ---------------------------------------------------------------------------
__global__ void zero_u32(unsigned* __restrict__ p, long n) {
  long i = (long)blockIdx.x * 256 + threadIdx.x;
  if (i < n) p[i] = 0u;
}

// f32 FC weights [M][K] -> f16 [M][K]
__global__ void cvt_weight(const float* __restrict__ src, _Float16* __restrict__ dst,
                           int n) {
  int gid = blockIdx.x * 256 + threadIdx.x;
  if (gid >= n) return;
  dst[gid] = (_Float16)src[gid];
}

// f32 conv weights [Cout][Cin][3][3] -> f16 [9][Cout][Cpad] (zero-pad channels)
__global__ void cvt_weight_conv(const float* __restrict__ src, _Float16* __restrict__ dst,
                                int Cout, int Cin, int Cpad) {
  int gid = blockIdx.x * 256 + threadIdx.x;
  if (gid >= 9 * Cout * Cpad) return;
  int c = gid % Cpad; int t = gid / Cpad;
  int m = t % Cout;   int j = t / Cout;
  dst[gid] = (c < Cin) ? (_Float16)src[(m * Cin + c) * 9 + j] : (_Float16)0.f;
}

// x f32 NCHW [B][3][224][224] -> padded NHWC f16 [B][226][226][32] (c>=3 stays 0)
__global__ void cvt_input(const float* __restrict__ x, _Float16* __restrict__ out) {
  int gid = blockIdx.x * 256 + threadIdx.x;
  const int total = BATCH * 3 * 224 * 224;
  if (gid >= total) return;
  int xx = gid % 224; int t = gid / 224;
  int yy = t % 224;   t /= 224;
  int c  = t % 3;     int b = t / 3;
  out[(((long)b * 226 + yy + 1) * 226 + xx + 1) * 32 + c] = (_Float16)x[gid];
}

// 2x2 stride-2 maxpool, padded NHWC f16, 8 channels per thread (b128 path)
__global__ void maxpool2(const _Float16* __restrict__ in, _Float16* __restrict__ out,
                         int C, int Hin, int Win) {
  int Hout = Hin >> 1, Wout = Win >> 1;
  int nc8 = C >> 3;
  long total = (long)BATCH * Hout * Wout * nc8;
  long gid = (long)blockIdx.x * 256 + threadIdx.x;
  if (gid >= total) return;
  int c8 = gid % nc8; long t = gid / nc8;
  int xx = t % Wout;  t /= Wout;
  int yy = t % Hout;  int b = t / Hout;
  const _Float16* p = in + (((long)b * (Hin + 2) + 2 * yy + 1) * (Win + 2) + 2 * xx + 1) * C + c8 * 8;
  long rs = (long)(Win + 2) * C;
  v8h a0 = *(const v8h*)p;
  v8h a1 = *(const v8h*)(p + C);
  v8h a2 = *(const v8h*)(p + rs);
  v8h a3 = *(const v8h*)(p + rs + C);
  v8h o;
#pragma unroll
  for (int e = 0; e < 8; ++e) {
    float m = fmaxf(fmaxf((float)a0[e], (float)a1[e]), fmaxf((float)a2[e], (float)a3[e]));
    o[e] = (_Float16)m;
  }
  *(v8h*)&out[(((long)b * (Hout + 2) + yy + 1) * (Wout + 2) + xx + 1) * C + c8 * 8] = o;
}

// adaptive pool: padded NHWC [B][16][16][256] (14x14 interior) -> feat f16 [4096][B]
__global__ void adpool(const _Float16* __restrict__ in, _Float16* __restrict__ feat) {
  int gid = blockIdx.x * 256 + threadIdx.x;
  if (gid >= 4096 * BATCH) return;
  int b = gid & 31; int f = gid >> 5;
  int c = f >> 4; int i = (f >> 2) & 3; int j = f & 3;
  const int starts[4] = {0, 3, 7, 10};
  int r0 = starts[i], c0 = starts[j];
  float s = 0.f;
#pragma unroll
  for (int r = 0; r < 4; ++r)
#pragma unroll
    for (int cc = 0; cc < 4; ++cc)
      s += (float)in[(((long)b * 16 + (1 + r0 + r)) * 16 + (1 + c0 + cc)) * 256 + c];
  feat[f * BATCH + b] = (_Float16)(s * (1.f / 16.f));
}

// ---------------------------------------------------------------------------
// 3x3 conv as 9 shifted GEMMs with WMMA, NHWC activations.
// A = weights f16 [9][Cout][Cpad]; B from padded NHWC acts [B][H+2][W+2][Cpad].
// Block: 256 thr (8 waves), tile 64(M=Cout) x 128(N=spatial).
// Software-pipelined: step i+1's global b128 loads issue while step i's WMMAs
// run; all global/LDS traffic on the A/B path is 16-byte vectors.
// Epilogue: bias + bn2d + relu, 8-channel packed stores to NHWC output.
// ---------------------------------------------------------------------------
__global__ __launch_bounds__(256) void conv_wmma(
    const _Float16* __restrict__ wA, const _Float16* __restrict__ actin,
    _Float16* __restrict__ actout,
    const float* __restrict__ bias, const float* __restrict__ gam,
    const float* __restrict__ bet, const float* __restrict__ mu,
    const float* __restrict__ var,
    int Cpad, int Cout, int H, int W) {
  __shared__ __align__(16) _Float16 sA[64 * 32];
  __shared__ __align__(16) _Float16 sB[128 * 32];
  const int tid = threadIdx.x;
  const int n0 = blockIdx.x * 128;
  const int m0 = blockIdx.y * 64;
  const int Hp = H + 2, Wp = W + 2;

  const int lane = tid & 31, wv = tid >> 5;
  const int r = lane & 15, hi = lane >> 4;
  const int wm = wv & 3, wn = wv >> 2;

  v8f acc[4] = {v8f{}, v8f{}, v8f{}, v8f{}};

  // B-fill geometry: this thread owns spatial n, channel chunk kof (0 or 16)
  const int fb_n = tid >> 1;
  const int fb_kof = (tid & 1) * 16;
  const int gn_f = n0 + fb_n;
  const int fb_x = gn_f % W;
  const int fb_t = gn_f / W;
  const int fb_y = fb_t % H;
  const int fb_b = fb_t / H;
  // A-fill geometry
  const int fa_m = tid >> 2;
  const int fa_kof = (tid & 3) * 8;
  const bool fa_ok = (m0 + fa_m) < Cout;

  // step = (tap j) * cper + (channel chunk); cper is a power of two (1,2,4,8)
  const int cper = Cpad >> 5;
  const int cbits = (cper == 1) ? 0 : (cper == 2) ? 1 : (cper == 4) ? 2 : 3;
  const int nsteps = 9 << cbits;

  const _Float16* brow = actin + (((long)fb_b * Hp + fb_y) * Wp + fb_x) * Cpad + fb_kof;
  const _Float16* abase = wA + (long)(m0 + fa_m) * Cpad + fa_kof;
  const long aTap = (long)Cout * Cpad;

  auto load_step = [&](int step, v8h& ra, v8h& rb0, v8h& rb1) {
    int j = step >> cbits;
    int c0 = (step & (cper - 1)) << 5;
    int ky = j / 3, kx = j - ky * 3;
    long boff = ((long)ky * Wp + kx) * Cpad + c0;
    rb0 = *(const v8h*)&brow[boff];
    rb1 = *(const v8h*)&brow[boff + 8];
    ra = v8h{};
    if (fa_ok) ra = *(const v8h*)&abase[(long)j * aTap + c0];
  };

  v8h ra, rb0, rb1;
  load_step(0, ra, rb0, rb1);

  for (int step = 0; step < nsteps; ++step) {
    __syncthreads();  // previous compute done reading LDS
    *(v8h*)&sA[fa_m * 32 + fa_kof]     = ra;
    *(v8h*)&sB[fb_n * 32 + fb_kof]     = rb0;
    *(v8h*)&sB[fb_n * 32 + fb_kof + 8] = rb1;
    v8h na = v8h{}, nb0 = v8h{}, nb1 = v8h{};
    if (step + 1 < nsteps) load_step(step + 1, na, nb0, nb1);  // in flight during WMMA
    __syncthreads();  // LDS tiles visible

    int ma = wm * 16 + r;
    v8h alo = *(const v8h*)&sA[ma * 32 + hi * 8];
    v8h ahi = *(const v8h*)&sA[ma * 32 + 16 + hi * 8];
    v16h a = __builtin_shufflevector(alo, ahi, 0, 1, 2, 3, 4, 5, 6, 7, 8, 9, 10, 11, 12, 13, 14, 15);
#pragma unroll
    for (int jj = 0; jj < 4; ++jj) {
      int nb = wn * 64 + jj * 16 + r;
      v8h blo = *(const v8h*)&sB[nb * 32 + hi * 8];
      v8h bh2 = *(const v8h*)&sB[nb * 32 + 16 + hi * 8];
      v16h bf = __builtin_shufflevector(blo, bh2, 0, 1, 2, 3, 4, 5, 6, 7, 8, 9, 10, 11, 12, 13, 14, 15);
      acc[jj] = __builtin_amdgcn_wmma_f32_16x16x32_f16(false, a, false, bf, (short)0, acc[jj],
                                                       false, false);
    }
    ra = na; rb0 = nb0; rb1 = nb1;
  }

  // Epilogue: relu(bn(z + bias)); lane owns 8 consecutive channels -> packed store
  const int mbase = m0 + wm * 16 + hi * 8;
  if (mbase < Cout) {
#pragma unroll
    for (int jj = 0; jj < 4; ++jj) {
      int gn = n0 + wn * 64 + jj * 16 + r;
      int xx = gn % W; int t = gn / W;
      int yy = t % H;  int bb = t / H;
      v8h ov;
#pragma unroll
      for (int v = 0; v < 8; ++v) {
        int gm = mbase + v;
        float sc = gam[gm] * rsqrtf(var[gm] + 1e-5f);
        float z = acc[jj][v] + bias[gm];
        z = (z - mu[gm]) * sc + bet[gm];
        ov[v] = (_Float16)fmaxf(z, 0.f);
      }
      *(v8h*)&actout[(((long)bb * Hp + (yy + 1)) * Wp + (xx + 1)) * Cout + mbase] = ov;
    }
  }
}

// ---------------------------------------------------------------------------
// FC layer with WMMA: out[m][b] = bn(relu(W[m][k] x[k][b] + bias[m])).
// A = W f16 [M][K], B = x f16 [K][32]. Block tile 64(M) x 32(N), 8 waves.
// ---------------------------------------------------------------------------
__global__ __launch_bounds__(256) void fc_wmma(
    const _Float16* __restrict__ wA, const _Float16* __restrict__ xin,
    _Float16* __restrict__ xout,
    const float* __restrict__ bias, const float* __restrict__ gam,
    const float* __restrict__ bet, const float* __restrict__ mu,
    const float* __restrict__ var, int M, int K) {
  __shared__ __align__(16) _Float16 sA[64 * 32];
  __shared__ __align__(16) _Float16 sB[32 * 32];
  const int tid = threadIdx.x;
  const int m0 = blockIdx.y * 64;
  const int lane = tid & 31, wv = tid >> 5;
  const int r = lane & 15, hi = lane >> 4;
  const int wm = wv & 3, wn = wv >> 2;

  v8f acc = v8f{};

  for (int k0 = 0; k0 < K; k0 += 32) {
    __syncthreads();
    {  // A: 64x32, one b128 per thread
      int m = tid >> 2;
      int kof = (tid & 3) * 8;
      *(v8h*)&sA[m * 32 + kof] = *(const v8h*)&wA[(long)(m0 + m) * K + k0 + kof];
    }
    {  // B: 32(n) x 32(k), gathered transposed from [K][32]
      int n = tid >> 3;
      int kof = (tid & 7) * 4;
#pragma unroll
      for (int e = 0; e < 4; ++e)
        sB[n * 32 + kof + e] = xin[(k0 + kof + e) * BATCH + n];
    }
    __syncthreads();

    int ma = wm * 16 + r;
    v8h alo = *(const v8h*)&sA[ma * 32 + hi * 8];
    v8h ahi = *(const v8h*)&sA[ma * 32 + 16 + hi * 8];
    v16h a = __builtin_shufflevector(alo, ahi, 0, 1, 2, 3, 4, 5, 6, 7, 8, 9, 10, 11, 12, 13, 14, 15);
    int nb = wn * 16 + r;
    v8h blo = *(const v8h*)&sB[nb * 32 + hi * 8];
    v8h bh2 = *(const v8h*)&sB[nb * 32 + 16 + hi * 8];
    v16h bf = __builtin_shufflevector(blo, bh2, 0, 1, 2, 3, 4, 5, 6, 7, 8, 9, 10, 11, 12, 13, 14, 15);
    acc = __builtin_amdgcn_wmma_f32_16x16x32_f16(false, a, false, bf, (short)0, acc, false, false);
  }

  int gn = wn * 16 + r;
#pragma unroll
  for (int v = 0; v < 8; ++v) {
    int gm = m0 + wm * 16 + hi * 8 + v;
    float z = acc[v] + bias[gm];
    z = fmaxf(z, 0.f);  // relu BEFORE bn (reference order for FC)
    float sc = gam[gm] * rsqrtf(var[gm] + 1e-5f);
    z = (z - mu[gm]) * sc + bet[gm];
    xout[gm * BATCH + gn] = (_Float16)z;
  }
}

// ---------------------------------------------------------------------------
// 8-qubit statevector circuit, one block per image, state in LDS.
// ---------------------------------------------------------------------------
struct cplx { float x, y; };
__device__ inline cplx cmul(cplx a, cplx b) { cplx r; r.x = a.x * b.x - a.y * b.y; r.y = a.x * b.y + a.y * b.x; return r; }
__device__ inline cplx cadd(cplx a, cplx b) { cplx r; r.x = a.x + b.x; r.y = a.y + b.y; return r; }
struct mat2 { cplx m[2][2]; };
__device__ inline mat2 mmul(const mat2& A, const mat2& B) {
  mat2 C;
#pragma unroll
  for (int i = 0; i < 2; ++i)
#pragma unroll
    for (int j = 0; j < 2; ++j)
      C.m[i][j] = cadd(cmul(A.m[i][0], B.m[0][j]), cmul(A.m[i][1], B.m[1][j]));
  return C;
}
__device__ inline mat2 g_rx(float t) {
  float c = cosf(0.5f * t), s = sinf(0.5f * t);
  mat2 g; g.m[0][0] = {c, 0}; g.m[0][1] = {0, -s}; g.m[1][0] = {0, -s}; g.m[1][1] = {c, 0}; return g;
}
__device__ inline mat2 g_ry(float t) {
  float c = cosf(0.5f * t), s = sinf(0.5f * t);
  mat2 g; g.m[0][0] = {c, 0}; g.m[0][1] = {-s, 0}; g.m[1][0] = {s, 0}; g.m[1][1] = {c, 0}; return g;
}
__device__ inline mat2 g_rz(float t) {
  float c = cosf(0.5f * t), s = sinf(0.5f * t);
  mat2 g; g.m[0][0] = {c, -s}; g.m[0][1] = {0, 0}; g.m[1][0] = {0, 0}; g.m[1][1] = {c, s}; return g;
}
__device__ inline mat2 g_h() {
  const float h = 0.70710678118654752f;
  mat2 g; g.m[0][0] = {h, 0}; g.m[0][1] = {h, 0}; g.m[1][0] = {h, 0}; g.m[1][1] = {-h, 0}; return g;
}
__device__ inline void apply_gate(cplx* st, const mat2& g, int q, int k) {
  __syncthreads();
  if (!(k & (1 << q))) {
    int k1 = k | (1 << q);
    cplx a0 = st[k], a1 = st[k1];
    st[k]  = cadd(cmul(g.m[0][0], a0), cmul(g.m[0][1], a1));
    st[k1] = cadd(cmul(g.m[1][0], a0), cmul(g.m[1][1], a1));
  }
  __syncthreads();
}

__global__ __launch_bounds__(256) void quantum_kernel(
    const _Float16* __restrict__ feat /*[128][B]*/, const float* __restrict__ theta,
    const float* __restrict__ phi, _Float16* __restrict__ qout /*[256][B]*/) {
  __shared__ cplx st[256];
  const int b = blockIdx.x;
  const int k = threadIdx.x;
  cplx init; init.x = (k == 0) ? 1.f : 0.f; init.y = 0.f;
  st[k] = init;
  __syncthreads();

  for (int q = 0; q < 8; ++q) {  // feature encoding: Ry(x) Rx(x) H
    float t = (float)feat[q * BATCH + b];
    mat2 g = mmul(g_ry(t), mmul(g_rx(t), g_h()));
    apply_gate(st, g, q, k);
  }
  for (int l = 0; l < 3; ++l) {
    for (int q = 0; q < 8; ++q) {  // U = Rz Ry Rx
      const float* th = theta + (l * 8 + q) * 3;
      mat2 g = mmul(g_rz(th[2]), mmul(g_ry(th[1]), g_rx(th[0])));
      apply_gate(st, g, q, k);
    }
    float ph = 0.f;  // diagonal ZZ phase
#pragma unroll
    for (int i = 0; i < 8; ++i)
#pragma unroll
      for (int j2 = i + 1; j2 < 8; ++j2) {
        float s = (((k >> i) & 1) ^ ((k >> j2) & 1)) ? 1.f : -1.f;
        ph += s * phi[l * 64 + i * 8 + j2];
      }
    ph *= 0.5f;
    cplx e; e.x = cosf(ph); e.y = sinf(ph);
    cplx a = st[k];
    st[k] = cmul(a, e);
    __syncthreads();
  }
  cplx a = st[k];
  qout[k * BATCH + b] = (_Float16)sqrtf(a.x * a.x + a.y * a.y);
}

// final 128 -> 1 + sigmoid
__global__ void final_fc(const _Float16* __restrict__ xin /*[128][B]*/,
                         const float* __restrict__ w, const float* __restrict__ bias,
                         float* __restrict__ out) {
  int b = threadIdx.x;
  if (b < BATCH) {
    float z = bias[0];
    for (int k = 0; k < 128; ++k) z += w[k] * (float)xin[k * BATCH + b];
    out[b] = 1.f / (1.f + expf(-z));
  }
}

// ---------------------------------------------------------------------------
// Host orchestration
// ---------------------------------------------------------------------------
extern "C" void kernel_launch(void* const* d_in, const int* in_sizes, int n_in,
                              void* d_out, int out_size, void* d_ws, size_t ws_size,
                              hipStream_t stream) {
  (void)in_sizes; (void)n_in; (void)out_size; (void)ws_size;
  char* ws = (char*)d_ws;
  size_t off = 0;
  auto alloc = [&](size_t bytes) -> char* {
    char* p = ws + off;
    off += (bytes + 255) & ~(size_t)255;
    return p;
  };

  const long BIG = 52301824L;  // halves: 32b * 226 * 226 * 32ch (NHWC padded)
  _Float16* bufA = (_Float16*)alloc(BIG * 2);
  _Float16* bufB = (_Float16*)alloc(BIG * 2);

  // conv layer configs (Cpad = input channels as stored, always >=32)
  const int Cpad_[8] = {32, 32, 32, 64, 64, 128, 128, 256};
  const int Cin_[8]  = {3, 32, 32, 64, 64, 128, 128, 256};
  const int Cout_[8] = {32, 32, 64, 64, 128, 128, 256, 256};
  const int Hc_[8]   = {224, 224, 112, 112, 56, 56, 28, 28};

  _Float16* wc[8];
  for (int i = 0; i < 8; ++i) wc[i] = (_Float16*)alloc((size_t)9 * Cout_[i] * Cpad_[i] * 2);
  // FC weights: (M,K) pairs
  const int fcM[5] = {512, 128, 512, 256, 128};
  const int fcK[5] = {4096, 512, 256, 512, 256};
  const int fcWidx[5] = {49, 51, 63, 65, 67};
  _Float16* wf[5];
  for (int i = 0; i < 5; ++i) wf[i] = (_Float16*)alloc((size_t)fcM[i] * fcK[i] * 2);

  _Float16* feat = (_Float16*)alloc(4096 * BATCH * 2);
  _Float16* f1   = (_Float16*)alloc(512 * BATCH * 2);
  _Float16* f2   = (_Float16*)alloc(128 * BATCH * 2);
  _Float16* qb   = (_Float16*)alloc(256 * BATCH * 2);
  _Float16* g1   = (_Float16*)alloc(512 * BATCH * 2);
  _Float16* g2   = (_Float16*)alloc(256 * BATCH * 2);
  _Float16* g3   = (_Float16*)alloc(128 * BATCH * 2);

  // --- weight conversion ---
  for (int i = 0; i < 8; ++i) {
    int n = 9 * Cout_[i] * Cpad_[i];
    cvt_weight_conv<<<(n + 255) / 256, 256, 0, stream>>>((const float*)d_in[1 + 2 * i], wc[i],
                                                         Cout_[i], Cin_[i], Cpad_[i]);
  }
  for (int i = 0; i < 5; ++i) {
    int n = fcM[i] * fcK[i];
    cvt_weight<<<(n + 255) / 256, 256, 0, stream>>>((const float*)d_in[fcWidx[i]], wf[i], n);
  }

  auto zero_region = [&](void* p, long halves) {
    long words = halves / 2;
    zero_u32<<<(unsigned)((words + 255) / 256), 256, 0, stream>>>((unsigned*)p, words);
  };
  auto run_conv = [&](int li, const _Float16* in, _Float16* out) {
    const float* bias = (const float*)d_in[2 + 2 * li];
    const float* g  = (const float*)d_in[17 + 4 * li + 0];
    const float* be = (const float*)d_in[17 + 4 * li + 1];
    const float* mu = (const float*)d_in[17 + 4 * li + 2];
    const float* va = (const float*)d_in[17 + 4 * li + 3];
    int H = Hc_[li];
    zero_region(out, (long)BATCH * (H + 2) * (H + 2) * Cout_[li]);
    dim3 grid((unsigned)((long)BATCH * H * H / 128), (unsigned)((Cout_[li] + 63) / 64));
    conv_wmma<<<grid, 256, 0, stream>>>(wc[li], in, out, bias, g, be, mu, va,
                                        Cpad_[li], Cout_[li], H, H);
  };
  auto run_pool = [&](const _Float16* in, _Float16* out, int C, int Hin) {
    int Ho = Hin / 2;
    zero_region(out, (long)BATCH * (Ho + 2) * (Ho + 2) * C);
    long total = (long)BATCH * Ho * Ho * (C / 8);
    maxpool2<<<(unsigned)((total + 255) / 256), 256, 0, stream>>>(in, out, C, Hin, Hin);
  };
  auto run_fc = [&](int fi, const _Float16* in, _Float16* out, int bidx, int bnidx) {
    dim3 grid(1, (unsigned)(fcM[fi] / 64));
    fc_wmma<<<grid, 256, 0, stream>>>(wf[fi], in, out,
                                      (const float*)d_in[bidx],
                                      (const float*)d_in[bnidx + 0], (const float*)d_in[bnidx + 1],
                                      (const float*)d_in[bnidx + 2], (const float*)d_in[bnidx + 3],
                                      fcM[fi], fcK[fi]);
  };

  // --- CNN pipeline (NHWC padded, ping-pong A/B) ---
  zero_region(bufA, (long)BATCH * 226 * 226 * 32);
  cvt_input<<<(BATCH * 3 * 224 * 224 + 255) / 256, 256, 0, stream>>>((const float*)d_in[0], bufA);

  run_conv(0, bufA, bufB);
  run_conv(1, bufB, bufA);
  run_pool(bufA, bufB, 32, 224);
  run_conv(2, bufB, bufA);
  run_conv(3, bufA, bufB);
  run_pool(bufB, bufA, 64, 112);
  run_conv(4, bufA, bufB);
  run_conv(5, bufB, bufA);
  run_pool(bufA, bufB, 128, 56);
  run_conv(6, bufB, bufA);
  run_conv(7, bufA, bufB);
  run_pool(bufB, bufA, 256, 28);

  adpool<<<(4096 * BATCH + 255) / 256, 256, 0, stream>>>(bufA, feat);

  // --- FC compression + quantum + classifier ---
  run_fc(0, feat, f1, 50, 53);  // 4096 -> 512
  run_fc(1, f1, f2, 52, 57);    // 512 -> 128
  quantum_kernel<<<BATCH, 256, 0, stream>>>(f2, (const float*)d_in[61],
                                            (const float*)d_in[62], qb);
  run_fc(2, qb, g1, 64, 71);    // 256 -> 512
  run_fc(3, g1, g2, 66, 75);    // 512 -> 256
  run_fc(4, g2, g3, 68, 79);    // 256 -> 128
  final_fc<<<1, 32, 0, stream>>>(g3, (const float*)d_in[69], (const float*)d_in[70],
                                 (float*)d_out);
}
